// SphericalAttentionBlock_8203387536061
// MI455X (gfx1250) — matrix operand order repaired and checked by reading, not compiled, not run on hardware
//
#include <hip/hip_runtime.h>
#include <math.h>

typedef _Float16 half_t;
typedef __attribute__((ext_vector_type(16))) _Float16 v16h;
typedef __attribute__((ext_vector_type(8)))  _Float16 v8h;
typedef __attribute__((ext_vector_type(8)))  float    v8f;
typedef __attribute__((ext_vector_type(4)))  float    v4f;

#define DEVI __device__ __forceinline__

constexpr int   NLAT  = 61;
constexpr int   NLON  = 120;
constexpr int   NSEG  = NLAT * NLON;   // 7320
constexpr int   NPAD  = 7328;          // 458 * 16
constexpr int   NT    = NPAD / 16;     // 458 column tiles
constexpr int   CCH   = 64;
constexpr int   DHD   = 16;
constexpr int   HID   = 128;
constexpr float SCALE = 0.25f;         // 1/sqrt(DH)
constexpr float LNEPS = 1e-6f;

// ---------------------------------------------------------------------------
// WMMA helpers (gfx1250 wave32, 16x16x32 f16 -> f32)
// ---------------------------------------------------------------------------
DEVI v8f wmma16(v16h a, v16h b, v8f c) {
  return __builtin_amdgcn_wmma_f32_16x16x32_f16(
      /*neg_a=*/false, a, /*neg_b=*/false, b,
      /*c_mod=*/(short)0, c, /*reuse_a=*/false, /*reuse_b=*/false);
}

// A fragment: W row-major [M][ld]; tile rows row0..row0+15, K slice kk..kk+31.
// lane%16 = M row; lane-half selects K half {0,16}; element s -> K=koff+s
// (16 consecutive K, matching load_b).  Two contiguous 16B loads.
DEVI v16h load_a(const half_t* __restrict__ w, int row0, int ld, int kk, int lane) {
  const half_t* p = w + (size_t)(row0 + (lane & 15)) * ld + kk + ((lane >> 4) << 4);
  v8h lo = *(const v8h*)(p);
  v8h hi = *(const v8h*)(p + 8);
  return __builtin_shufflevector(lo, hi, 0,1,2,3,4,5,6,7,8,9,10,11,12,13,14,15);
}

// B fragment: activation point-major [n][ld]; lane%16 = column n, lane-half
// selects K half {0,16}; 16 consecutive K per lane -> one contiguous 32B load.
DEVI v16h load_b(const half_t* __restrict__ act, int n0, int ld, int kk, int lane) {
  const half_t* p = act + (size_t)(n0 + (lane & 15)) * ld + kk + ((lane >> 4) << 4);
  return *(const v16h*)p;
}

// ---------------------------------------------------------------------------
// Weight conversion f32 -> f16
// ---------------------------------------------------------------------------
__global__ void cvt_kernel(const float* __restrict__ s, half_t* __restrict__ d, int n) {
  int i = blockIdx.x * blockDim.x + threadIdx.x;
  if (i < n) d[i] = (half_t)s[i];
}

// ---------------------------------------------------------------------------
// Channel LayerNorm; src element (c,n) at src[c*sC + n*sN]; out f16 [n][64].
// Pad columns written as zero.
// ---------------------------------------------------------------------------
__global__ void ln_kernel(const float* __restrict__ src, int sC, int sN,
                          const float* __restrict__ g, const float* __restrict__ bb,
                          half_t* __restrict__ dst) {
  int n = blockIdx.x * blockDim.x + threadIdx.x;
  if (n >= NPAD) return;
  half_t* d = dst + (size_t)n * CCH;
  if (n >= NSEG) {
    #pragma unroll
    for (int c = 0; c < CCH; ++c) d[c] = (half_t)0.f;
    return;
  }
  float vals[CCH];
  float mean = 0.f;
  #pragma unroll
  for (int c = 0; c < CCH; ++c) { float t = src[(size_t)c * sC + (size_t)n * sN]; vals[c] = t; mean += t; }
  mean *= (1.f / CCH);
  float var = 0.f;
  #pragma unroll
  for (int c = 0; c < CCH; ++c) { float t = vals[c] - mean; var += t * t; }
  var *= (1.f / CCH);
  float rs = rsqrtf(var + LNEPS);
  #pragma unroll
  for (int c = 0; c < CCH; ++c) d[c] = (half_t)((vals[c] - mean) * rs * g[c] + bb[c]);
}

// ---------------------------------------------------------------------------
// Fused Q/K/V projection: q[n][o] = sum_c wq[o][c] * h0[n][c]  (f32 out)
// One wave per 16-column tile; B fragment reused across 12 WMMA accumulators.
// ---------------------------------------------------------------------------
__global__ void qkv_gemm_kernel(const half_t* __restrict__ h0,
                                const half_t* __restrict__ wq16,
                                const half_t* __restrict__ wk16,
                                const half_t* __restrict__ wv16,
                                float* __restrict__ q, float* __restrict__ k,
                                float* __restrict__ v) {
  const int wave = threadIdx.x >> 5;
  const int lane = threadIdx.x & 31;
  const int tile = blockIdx.x * 4 + wave;
  if (tile >= NT) return;
  const int n0 = tile * 16;

  v8f aq[4] = {}, ak[4] = {}, av[4] = {};
  #pragma unroll
  for (int kk = 0; kk < CCH; kk += 32) {
    v16h b = load_b(h0, n0, CCH, kk, lane);
    #pragma unroll
    for (int mt = 0; mt < 4; ++mt) {
      v16h a;
      a = load_a(wq16, mt * 16, CCH, kk, lane); aq[mt] = wmma16(a, b, aq[mt]);
      a = load_a(wk16, mt * 16, CCH, kk, lane); ak[mt] = wmma16(a, b, ak[mt]);
      a = load_a(wv16, mt * 16, CCH, kk, lane); av[mt] = wmma16(a, b, av[mt]);
    }
  }
  const int col = n0 + (lane & 15);
  if (col >= NSEG) return;
  const int mb = (lane >> 4) << 3;
  #pragma unroll
  for (int mt = 0; mt < 4; ++mt)
    #pragma unroll
    for (int r = 0; r < 8; ++r) {
      int M = mt * 16 + mb + r;
      q[(size_t)col * CCH + M] = aq[mt][r];
      k[(size_t)col * CCH + M] = ak[mt][r];
      v[(size_t)col * CCH + M] = av[mt][r];
    }
}

// ---------------------------------------------------------------------------
// Neighborhood attention with quadrature-weighted online softmax.
// One block (128 thr) per output point; one wave per head. The entry range
// for the output's latitude ring is found by binary search over out_idx
// (entries grouped by latitude: out_idx[e*NLON] / NLON is nondecreasing).
// ---------------------------------------------------------------------------
__global__ void attn_kernel(const float* __restrict__ qb, const float* __restrict__ kb,
                            const float* __restrict__ vb,
                            const int* __restrict__ out_idx, const int* __restrict__ in_idx,
                            const float* __restrict__ quad_w, int E,
                            half_t* __restrict__ atth) {
  const int n    = blockIdx.x;
  const int h    = threadIdx.x >> 5;
  const int lane = threadIdx.x & 31;
  const int lat  = n / NLON;
  const int w    = n - lat * NLON;

  int lo = 0, hi = E;
  while (lo < hi) { int mid = (lo + hi) >> 1; if (out_idx[(size_t)mid * NLON] / NLON < lat) lo = mid + 1; else hi = mid; }
  const int eLo = lo;
  hi = E;
  while (lo < hi) { int mid = (lo + hi) >> 1; if (out_idx[(size_t)mid * NLON] / NLON < lat + 1) lo = mid + 1; else hi = mid; }
  const int eHi = lo;

  const float* qp = qb + (size_t)n * CCH + h * DHD;
  v4f q0 = *(const v4f*)(qp + 0), q1 = *(const v4f*)(qp + 4);
  v4f q2 = *(const v4f*)(qp + 8), q3 = *(const v4f*)(qp + 12);

  float m = -1e30f, den = 0.f;
  float num[DHD];
  #pragma unroll
  for (int d = 0; d < DHD; ++d) num[d] = 0.f;

  for (int e = eLo + lane; e < eHi; e += 32) {
    const int ii = in_idx[(size_t)e * NLON + w];
    const float qw = quad_w[ii];
    const float* kp = kb + (size_t)ii * CCH + h * DHD;
    v4f k0 = *(const v4f*)(kp + 0), k1 = *(const v4f*)(kp + 4);
    v4f k2 = *(const v4f*)(kp + 8), k3 = *(const v4f*)(kp + 12);
    float s = 0.f;
    #pragma unroll
    for (int j = 0; j < 4; ++j) s += q0[j]*k0[j] + q1[j]*k1[j] + q2[j]*k2[j] + q3[j]*k3[j];
    s *= SCALE;
    const float* vp = vb + (size_t)ii * CCH + h * DHD;
    v4f v0 = *(const v4f*)(vp + 0), v1 = *(const v4f*)(vp + 4);
    v4f v2 = *(const v4f*)(vp + 8), v3 = *(const v4f*)(vp + 12);

    float nm = fmaxf(m, s);
    float c  = __expf(m - nm);
    float wg = __expf(s - nm) * qw;
    den = den * c + wg;
    #pragma unroll
    for (int j = 0; j < 4; ++j) {
      num[j +  0] = num[j +  0] * c + wg * v0[j];
      num[j +  4] = num[j +  4] * c + wg * v1[j];
      num[j +  8] = num[j +  8] * c + wg * v2[j];
      num[j + 12] = num[j + 12] * c + wg * v3[j];
    }
    m = nm;
  }

  // Butterfly combine of online-softmax states across the 32 lanes.
  #pragma unroll
  for (int off = 16; off > 0; off >>= 1) {
    float m2 = __shfl_xor(m, off, 32);
    float d2 = __shfl_xor(den, off, 32);
    float M  = fmaxf(m, m2);
    float c1 = __expf(m - M), c2 = __expf(m2 - M);
    den = den * c1 + d2 * c2;
    #pragma unroll
    for (int d = 0; d < DHD; ++d) {
      float t = __shfl_xor(num[d], off, 32);
      num[d] = num[d] * c1 + t * c2;
    }
    m = M;
  }

  if (lane == 0) {
    float inv = 1.f / den;
    #pragma unroll
    for (int d = 0; d < DHD; ++d)
      atth[(size_t)n * CCH + h * DHD + d] = (half_t)(num[d] * inv);
  }
}

// ---------------------------------------------------------------------------
// Output projection + residual: x1[n][o] = sum_c wo[o][c]*att[n][c] + x[o][n]
// ---------------------------------------------------------------------------
__global__ void wo_gemm_kernel(const half_t* __restrict__ atth,
                               const half_t* __restrict__ wo16,
                               const float* __restrict__ x,
                               float* __restrict__ x1) {
  const int wave = threadIdx.x >> 5;
  const int lane = threadIdx.x & 31;
  const int tile = blockIdx.x * 4 + wave;
  if (tile >= NT) return;
  const int n0 = tile * 16;

  v8f acc[4] = {};
  #pragma unroll
  for (int kk = 0; kk < CCH; kk += 32) {
    v16h b = load_b(atth, n0, CCH, kk, lane);
    #pragma unroll
    for (int mt = 0; mt < 4; ++mt) {
      v16h a = load_a(wo16, mt * 16, CCH, kk, lane);
      acc[mt] = wmma16(a, b, acc[mt]);
    }
  }
  const int col = n0 + (lane & 15);
  if (col >= NSEG) return;
  const int mb = (lane >> 4) << 3;
  #pragma unroll
  for (int mt = 0; mt < 4; ++mt)
    #pragma unroll
    for (int r = 0; r < 8; ++r) {
      int M = mt * 16 + mb + r;
      x1[(size_t)col * CCH + M] = acc[mt][r] + x[(size_t)M * NSEG + col];
    }
}

// ---------------------------------------------------------------------------
// MLP layer 1: hid[n][f] = gelu(sum_c w1[f][c]*h1[n][c] + b1[f])  (f16 out)
// ---------------------------------------------------------------------------
__global__ void mlp1_kernel(const half_t* __restrict__ h1,
                            const half_t* __restrict__ w1h,
                            const float* __restrict__ b1,
                            half_t* __restrict__ hid) {
  const int wave = threadIdx.x >> 5;
  const int lane = threadIdx.x & 31;
  const int tile = blockIdx.x * 4 + wave;
  if (tile >= NT) return;
  const int n0 = tile * 16;

  v8f acc[8] = {};
  #pragma unroll
  for (int kk = 0; kk < CCH; kk += 32) {
    v16h b = load_b(h1, n0, CCH, kk, lane);
    #pragma unroll
    for (int mt = 0; mt < 8; ++mt) {
      v16h a = load_a(w1h, mt * 16, CCH, kk, lane);
      acc[mt] = wmma16(a, b, acc[mt]);
    }
  }
  const int col = n0 + (lane & 15);
  if (col >= NSEG) return;
  const int mb = (lane >> 4) << 3;
  #pragma unroll
  for (int mt = 0; mt < 8; ++mt)
    #pragma unroll
    for (int r = 0; r < 8; ++r) {
      int M = mt * 16 + mb + r;
      float t = acc[mt][r] + b1[M];
      float g = 0.5f * t * (1.0f + erff(t * 0.70710678118654752f));  // exact GELU
      hid[(size_t)col * HID + M] = (half_t)g;
    }
}

// ---------------------------------------------------------------------------
// MLP layer 2 + bias + residual, output in reference layout [C][NSEG] f32.
// ---------------------------------------------------------------------------
__global__ void mlp2_kernel(const half_t* __restrict__ hid,
                            const half_t* __restrict__ w2h,
                            const float* __restrict__ b2,
                            const float* __restrict__ x1,
                            float* __restrict__ out) {
  const int wave = threadIdx.x >> 5;
  const int lane = threadIdx.x & 31;
  const int tile = blockIdx.x * 4 + wave;
  if (tile >= NT) return;
  const int n0 = tile * 16;

  v8f acc[4] = {};
  #pragma unroll
  for (int kk = 0; kk < HID; kk += 32) {
    v16h b = load_b(hid, n0, HID, kk, lane);
    #pragma unroll
    for (int mt = 0; mt < 4; ++mt) {
      v16h a = load_a(w2h, mt * 16, HID, kk, lane);
      acc[mt] = wmma16(a, b, acc[mt]);
    }
  }
  const int col = n0 + (lane & 15);
  if (col >= NSEG) return;
  const int mb = (lane >> 4) << 3;
  #pragma unroll
  for (int mt = 0; mt < 4; ++mt)
    #pragma unroll
    for (int r = 0; r < 8; ++r) {
      int M = mt * 16 + mb + r;
      out[(size_t)M * NSEG + col] = acc[mt][r] + b2[M] + x1[(size_t)col * CCH + M];
    }
}

// ---------------------------------------------------------------------------
extern "C" void kernel_launch(void* const* d_in, const int* in_sizes, int n_in,
                              void* d_out, int out_size, void* d_ws, size_t ws_size,
                              hipStream_t stream) {
  const float* x     = (const float*)d_in[0];
  const float* wq    = (const float*)d_in[1];
  const float* wk    = (const float*)d_in[2];
  const float* wv    = (const float*)d_in[3];
  const float* wo    = (const float*)d_in[4];
  const float* w1    = (const float*)d_in[5];
  const float* b1    = (const float*)d_in[6];
  const float* w2    = (const float*)d_in[7];
  const float* b2    = (const float*)d_in[8];
  const float* ln0_g = (const float*)d_in[9];
  const float* ln0_b = (const float*)d_in[10];
  const float* ln1_g = (const float*)d_in[11];
  const float* ln1_b = (const float*)d_in[12];
  const float* quadw = (const float*)d_in[13];
  const int*   oidx  = (const int*)d_in[14];
  const int*   iidx  = (const int*)d_in[15];
  const int    E     = in_sizes[14] / NLON;   // reference-column entries

  // Workspace carve-out (256B aligned slabs).
  char* p = (char*)d_ws;
  auto take = [&](size_t bytes) -> void* {
    void* r = (void*)p;
    p += (bytes + 255) & ~(size_t)255;
    return r;
  };
  half_t* h0h  = (half_t*)take((size_t)NPAD * CCH * sizeof(half_t));
  float*  qb   = (float*) take((size_t)NPAD * CCH * sizeof(float));
  float*  kb   = (float*) take((size_t)NPAD * CCH * sizeof(float));
  float*  vb   = (float*) take((size_t)NPAD * CCH * sizeof(float));
  half_t* atth = (half_t*)take((size_t)NPAD * CCH * sizeof(half_t));
  float*  x1   = (float*) take((size_t)NPAD * CCH * sizeof(float));
  half_t* h1h  = (half_t*)take((size_t)NPAD * CCH * sizeof(half_t));
  half_t* hidh = (half_t*)take((size_t)NPAD * HID * sizeof(half_t));
  half_t* wq16 = (half_t*)take(64 * 64 * sizeof(half_t));
  half_t* wk16 = (half_t*)take(64 * 64 * sizeof(half_t));
  half_t* wv16 = (half_t*)take(64 * 64 * sizeof(half_t));
  half_t* wo16 = (half_t*)take(64 * 64 * sizeof(half_t));
  half_t* w116 = (half_t*)take(128 * 64 * sizeof(half_t));
  half_t* w216 = (half_t*)take(64 * 128 * sizeof(half_t));

  // 1) Weight conversion.
  cvt_kernel<<<(4096 + 255) / 256, 256, 0, stream>>>(wq, wq16, 4096);
  cvt_kernel<<<(4096 + 255) / 256, 256, 0, stream>>>(wk, wk16, 4096);
  cvt_kernel<<<(4096 + 255) / 256, 256, 0, stream>>>(wv, wv16, 4096);
  cvt_kernel<<<(4096 + 255) / 256, 256, 0, stream>>>(wo, wo16, 4096);
  cvt_kernel<<<(8192 + 255) / 256, 256, 0, stream>>>(w1, w116, 8192);
  cvt_kernel<<<(8192 + 255) / 256, 256, 0, stream>>>(w2, w216, 8192);

  const int lnGrid   = (NPAD + 255) / 256;
  const int gemmGrid = (NT + 3) / 4;   // 4 waves / block, 1 column tile / wave

  // 2) LN0 (x is [C][NSEG] c-major)
  ln_kernel<<<lnGrid, 256, 0, stream>>>(x, NSEG, 1, ln0_g, ln0_b, h0h);
  // 3) Q/K/V projections (WMMA)
  qkv_gemm_kernel<<<gemmGrid, 128, 0, stream>>>(h0h, wq16, wk16, wv16, qb, kb, vb);
  // 4) Neighborhood attention, online softmax
  attn_kernel<<<NSEG, 128, 0, stream>>>(qb, kb, vb, oidx, iidx, quadw, E, atth);
  // 5) Wo projection + residual (WMMA)
  wo_gemm_kernel<<<gemmGrid, 128, 0, stream>>>(atth, wo16, x, x1);
  // 6) LN1 (x1 is [n][C] point-major)
  ln_kernel<<<lnGrid, 256, 0, stream>>>(x1, 1, CCH, ln1_g, ln1_b, h1h);
  // 7) MLP layer 1 + GELU (WMMA)
  mlp1_kernel<<<gemmGrid, 128, 0, stream>>>(h1h, w116, b1, hidh);
  // 8) MLP layer 2 + bias + residual -> d_out (WMMA)
  mlp2_kernel<<<gemmGrid, 128, 0, stream>>>(hidh, w216, b2, x1, (float*)d_out);
}
